// GCN_3435973837349
// MI455X (gfx1250) — matrix operand order repaired and checked by reading, not compiled, not run on hardware
//
#include <hip/hip_runtime.h>
#include <math.h>

typedef _Float16 half_t;
typedef _Float16 v16h __attribute__((ext_vector_type(16)));
typedef _Float16 v8h  __attribute__((ext_vector_type(8)));
typedef float    v8f  __attribute__((ext_vector_type(8)));

#define NODES 21
#define GPB   16           // graphs per block -> 336 rows = 21 exact 16-row M tiles
#define ROWS  (NODES*GPB)  // 336
#define F1 64
#define F2 128
#define NC 29
#define BN_EPS 1e-5f

__global__ __launch_bounds__(256) void gcn_fused(
    const float* __restrict__ x,
    const float* __restrict__ W1, const float* __restrict__ b1,
    const float* __restrict__ g1, const float* __restrict__ be1,
    const float* __restrict__ m1, const float* __restrict__ v1,
    const float* __restrict__ W2, const float* __restrict__ b2,
    const float* __restrict__ g2, const float* __restrict__ be2,
    const float* __restrict__ m2, const float* __restrict__ v2,
    const float* __restrict__ W3, const float* __restrict__ b3,
    float* __restrict__ out)
{
  __shared__ float Zs[NODES][NODES];      // normalized adjacency (same for all graphs)
  __shared__ float ps[NODES];             // pooling weights p = (1/21) * Z^T * 1
  __shared__ float Ms[ROWS][3];           // M = Z * X per graph
  __shared__ float s1[F1], t1[F1], s2[F2], t2[F2];  // folded BN affines
  __shared__ float W1s[3][F1];
  __shared__ __align__(16) half_t H1p[ROWS + 1][F1]; // +1 dump row: unconditional stores
  __shared__ float Us[GPB][F2];           // u[g] = sum_n p[n] * H2[g][n][:]
  __shared__ float Os[GPB][32];           // pre-softmax logits
  __shared__ int   rowG[ROWS];            // row -> graph index (kills div-by-21 in hot loop)
  __shared__ float rowP[ROWS];            // row -> ps[row % 21]

  const int tid  = threadIdx.x;
  const int lane = tid & 31;
  const int wave = tid >> 5;
  const int halfSel = (lane < 16) ? 0 : 1;
  const int lc = lane & 15;
  const long rowBase = (long)blockIdx.x * ROWS;

  // ---------------- phase A: zero accumulators ----------------
  for (int i = tid; i < NODES*NODES; i += 256) ((float*)Zs)[i] = 0.f;
  for (int i = tid; i < GPB*F2;      i += 256) ((float*)Us)[i] = 0.f;
  __syncthreads();

  // ---------------- phase B: build Z + constants ----------------
  if (tid == 0) {
    const int er[23] = {0,1,2,3,0,5,6,7,0,9,10,11,0,13,14,15,0,17,18,19,5,9,13};
    const int ec[23] = {1,2,3,4,5,6,7,8,9,10,11,12,13,14,15,16,17,18,19,20,9,13,17};
    float dinv[NODES]; int deg[NODES];
    for (int n = 0; n < NODES; n++) deg[n] = 1;            // self loop
    for (int e = 0; e < 23; e++)    deg[ec[e]]++;          // in-degree on col
    for (int n = 0; n < NODES; n++) dinv[n] = rsqrtf((float)deg[n]);
    for (int e = 0; e < 23; e++)
      Zs[ec[e]][er[e]] += dinv[er[e]] * dinv[ec[e]];       // agg[col] += h[row]*...
    for (int n = 0; n < NODES; n++) Zs[n][n] += dinv[n] * dinv[n];
  }
  for (int i = tid; i < 3*F1; i += 256) ((float*)W1s)[i] = W1[i];
  if (tid >= 64 && tid < 128) {
    int c = tid - 64;
    float s = g1[c] * rsqrtf(v1[c] + BN_EPS);
    s1[c] = s; t1[c] = (b1[c] - m1[c]) * s + be1[c];
  }
  if (tid >= 128) {
    int c = tid - 128;
    float s = g2[c] * rsqrtf(v2[c] + BN_EPS);
    s2[c] = s; t2[c] = (b2[c] - m2[c]) * s + be2[c];
  }
  __syncthreads();

  // ---------------- phase C: ps, row tables, M = Z * X ----------------
  if (tid < NODES) {
    float a = 0.f;
    for (int r = 0; r < NODES; r++) a += Zs[r][tid];
    ps[tid] = a * (1.f / NODES);
  }
  __syncthreads();   // ps needed for rowP below
  for (int i = tid; i < ROWS; i += 256) {
    int g = i / NODES;
    rowG[i] = g;
    rowP[i] = ps[i - g*NODES];
  }
  for (int i = tid; i < ROWS*3; i += 256) {
    int node = i / 3, d = i - node*3;
    int g = node / NODES, n = node - g*NODES;
    const float* xg = x + (rowBase + (long)g*NODES)*3 + d;
    float acc = 0.f;
    #pragma unroll
    for (int m = 0; m < NODES; m++) acc += Zs[n][m] * xg[m*3];
    Ms[node][d] = acc;
  }
  __syncthreads();

  // ---------------- pass 2: H1' = Z * H1 via WMMA (per graph) ----------------
  // A layout (16-bit 16x32): lane holds row lane%16; halfs i -> K = (i<8?i:i+8)+8*halfSel
  v16h aZ0, aZ1;
  #pragma unroll
  for (int i = 0; i < 16; i++) {
    int k = ((i < 8) ? i : i + 8) + halfSel*8;
    int r1 = 16 + lc;
    aZ0[i] = (half_t)((k < NODES) ? Zs[lc][k] : 0.f);
    aZ1[i] = (half_t)((r1 < NODES && k < NODES) ? Zs[r1][k] : 0.f);
  }
  for (int gg = 0; gg < 2; gg++) {
    int g = wave*2 + gg;                       // 8 waves x 2 graphs = 16 graphs
    for (int j = 0; j < 4; j++) {              // 4 feature tiles of 16 (F1=64)
      int c = j*16 + lc;
      float sc = s1[c], tc = t1[c];
      float w0 = W1s[0][c], w1 = W1s[1][c], w2c = W1s[2][c];
      v16h b;                                  // B layout: lane=col c, halfs i -> K=(i<8?i:i+8)+8*halfSel
      #pragma unroll
      for (int i = 0; i < 16; i++) {
        int k = ((i < 8) ? i : i + 8) + halfSel*8;
        float hv = 0.f;
        if (k < NODES) {                       // K-pad rows forced to exact 0
          int node = g*NODES + k;
          float dot = Ms[node][0]*w0 + Ms[node][1]*w1 + Ms[node][2]*w2c;
          hv = dot*sc + tc;
          hv = hv > 0.f ? hv : 0.f;            // relu(bn1(...))
        }
        b[i] = (half_t)hv;
      }
      v8f c0 = {};
      c0 = __builtin_amdgcn_wmma_f32_16x16x32_f16(false, aZ0, false, b, (short)0, c0, false, false);
      v8f c1 = {};
      c1 = __builtin_amdgcn_wmma_f32_16x16x32_f16(false, aZ1, false, b, (short)0, c1, false, false);
      #pragma unroll
      for (int r = 0; r < 8; r++) {            // C layout: lane=col, vgpr r = row r + 8*halfSel
        int row0 = r + halfSel*8;
        int row1 = 16 + row0;
        // unconditional stores: out-of-range rows go to the dump row (no exec masking)
        int idx1 = (row1 < NODES) ? (g*NODES + row1) : ROWS;
        H1p[g*NODES + row0][c] = (half_t)c0[r];          // rows 0..15 always valid
        H1p[idx1][c]           = (half_t)c1[r];
      }
    }
  }
  __syncthreads();

  // ---------------- pass 3: H1'(336x64) * W2(64x128) via WMMA; fold pool into U ------
  const int cB = wave*16 + lc;                 // this lane's output column (0..127)
  v16h bW0, bW1;                               // W2 B-tiles, K-steps 0/1, kept in regs
  #pragma unroll
  for (int i = 0; i < 16; i++) {
    int k0 = 0  + ((i < 8) ? i : i + 8) + halfSel*8;
    int k1 = 32 + ((i < 8) ? i : i + 8) + halfSel*8;
    bW0[i] = (half_t)W2[k0*F2 + cB];
    bW1[i] = (half_t)W2[k1*F2 + cB];
  }
  const float sC = s2[cB], tC = t2[cB];
  for (int m = 0; m < ROWS/16; m++) {          // 21 M tiles
    int node = m*16 + lc;
    int off  = halfSel*8;
    union { v16h v; v8h h[2]; } a0, a1;        // two contiguous 16B LDS chunks per K-step
    a0.h[0] = *(const v8h*)&H1p[node][off];
    a0.h[1] = *(const v8h*)&H1p[node][off + 16];
    a1.h[0] = *(const v8h*)&H1p[node][32 + off];
    a1.h[1] = *(const v8h*)&H1p[node][32 + off + 16];
    v8f acc = {};
    acc = __builtin_amdgcn_wmma_f32_16x16x32_f16(false, a0.v, false, bW0, (short)0, acc, false, false);
    acc = __builtin_amdgcn_wmma_f32_16x16x32_f16(false, a1.v, false, bW1, (short)0, acc, false, false);
    #pragma unroll
    for (int r = 0; r < 8; r++) {
      int row = m*16 + r + halfSel*8;          // 0..335, all valid
      float h = acc[r]*sC + tC;
      h = h > 0.f ? h : 0.f;                   // H2 = relu(bn2(...))
      atomicAdd(&Us[rowG[row]][cB], rowP[row] * h);   // pooled contribution (ds_add_f32)
    }
  }
  __syncthreads();

  // ---------------- pass 4: O = U * W3 + b3 (16x128 @ 128x29, tiny) ----------------
  for (int o = tid; o < GPB*NC; o += 256) {
    int g = o / NC, c = o - g*NC;
    float acc = b3[c];
    #pragma unroll 4
    for (int k = 0; k < F2; k++) acc += Us[g][k] * W3[k*NC + c];
    Os[g][c] = acc;
  }
  __syncthreads();

  // ---------------- pass 5: log_softmax + store ----------------
  if (tid < GPB) {
    int g = tid;
    float mx = Os[g][0];
    for (int c = 1; c < NC; c++) mx = fmaxf(mx, Os[g][c]);
    float se = 0.f;
    for (int c = 0; c < NC; c++) se += __expf(Os[g][c] - mx);
    float lse = mx + __logf(se);
    float* op = out + ((long)blockIdx.x*GPB + g)*NC;
    for (int c = 0; c < NC; c++) op[c] = Os[g][c] - lse;
  }
}

extern "C" void kernel_launch(void* const* d_in, const int* in_sizes, int n_in,
                              void* d_out, int out_size, void* d_ws, size_t ws_size,
                              hipStream_t stream) {
  const float* x   = (const float*)d_in[0];
  const float* W1  = (const float*)d_in[1];
  const float* b1  = (const float*)d_in[2];
  const float* g1  = (const float*)d_in[3];
  const float* be1 = (const float*)d_in[4];
  const float* m1  = (const float*)d_in[5];
  const float* v1  = (const float*)d_in[6];
  const float* W2  = (const float*)d_in[7];
  const float* b2  = (const float*)d_in[8];
  const float* g2  = (const float*)d_in[9];
  const float* be2 = (const float*)d_in[10];
  const float* m2  = (const float*)d_in[11];
  const float* v2  = (const float*)d_in[12];
  const float* W3  = (const float*)d_in[13];
  const float* b3  = (const float*)d_in[14];

  int nGraphs = in_sizes[0] / (NODES * 3);   // 50000
  int nBlocks = nGraphs / GPB;               // 3125 (50000 divisible by 16)

  gcn_fused<<<nBlocks, 256, 0, stream>>>(x, W1, b1, g1, be1, m1, v1,
                                         W2, b2, g2, be2, m2, v2,
                                         W3, b3, (float*)d_out);
}
